// Attention_15126874817221
// MI455X (gfx1250) — matrix operand order repaired and checked by reading, not compiled, not run on hardware
//
#include <hip/hip_runtime.h>
#include <hip/hip_bf16.h>

// ---------------------------------------------------------------------------
// Problem constants (B=4, S=4096, E=512, fp32 in/out)
// ---------------------------------------------------------------------------
#define Bn 4
#define Sn 4096
#define En 512
#define BS (Bn * Sn)          // 16384 total rows
#define KT 64                 // key tile per flash iteration

typedef __attribute__((ext_vector_type(16))) __bf16 v16bf;
typedef __attribute__((ext_vector_type(8)))  float  v8f;

union Frag16 {                 // 32 bytes: one WMMA bf16 A/B operand per lane
    v16bf v;
    uint4 q[2];
    __bf16 e[16];
};
union Acc8 {                   // 32 bytes: one WMMA f32 C/D tile per lane
    v8f   v;
    float f[8];
};

__device__ inline v8f wmma_bf16(v16bf a, v16bf b, v8f c) {
    // D = A(16x32 bf16) * B(32x16 bf16) + C(16x16 f32)
    return __builtin_amdgcn_wmma_f32_16x16x32_bf16(
        false, a, false, b, (short)0, c, false, false);
}

// ---------------------------------------------------------------------------
// Pass 0: elementwise f32 -> bf16 conversion (hoists all cvt out of GEMMs).
// Each thread converts 8 elements: 2x b128 load, 1x b128 store.
// ---------------------------------------------------------------------------
__global__ __launch_bounds__(256, 1)
void cvt_bf16_kernel(const float* __restrict__ src, __bf16* __restrict__ dst)
{
    const int i = blockIdx.x * 256 + threadIdx.x;       // group-of-8 index
    const float4* s = (const float4*)src + (size_t)i * 2;
    float4 a = s[0], b = s[1];
    union { uint4 q; __bf16 e[8]; } o;
    o.e[0] = (__bf16)a.x; o.e[1] = (__bf16)a.y;
    o.e[2] = (__bf16)a.z; o.e[3] = (__bf16)a.w;
    o.e[4] = (__bf16)b.x; o.e[5] = (__bf16)b.y;
    o.e[6] = (__bf16)b.z; o.e[7] = (__bf16)b.w;
    ((uint4*)dst)[i] = o.q;
}

// ---------------------------------------------------------------------------
// Pass 1: QKV projection GEMM on pre-converted bf16 operands.
//   q = x @ W^T + b;  Q pre-scaled by 1/sqrt(E).
//   Q, K stored row-major bf16 [BS, E];  V stored transposed Vt[b][e][s].
// Each wave computes one 16(M) x 64(N) tile; 8 waves / block; grid.y = matrix.
// ---------------------------------------------------------------------------
__global__ __launch_bounds__(256, 1)
void qkv_proj_kernel(const __bf16* __restrict__ Xb,
                     const __bf16* __restrict__ Wb,   // [3][E][E] bf16
                     const float* __restrict__ bq,
                     const float* __restrict__ bk,
                     const float* __restrict__ bv,
                     __bf16* __restrict__ Qb,
                     __bf16* __restrict__ Kb,
                     __bf16* __restrict__ Vt)
{
    const int mat  = blockIdx.y;                       // 0=Q, 1=K, 2=V
    const __bf16* W   = Wb + (size_t)mat * En * En;
    const float* bias = (mat == 0) ? bq : (mat == 1) ? bk : bv;

    const int wave = threadIdx.x >> 5;
    const int lane = threadIdx.x & 31;
    const int lrow = lane & 15;
    const bool hi  = lane >= 16;

    const int task  = blockIdx.x * 8 + wave;           // 8192 tasks per matrix
    const int mtile = task >> 3;                       // 1024 M-tiles
    const int ng    = task & 7;                        // 8 N-groups of 64
    const int m0    = mtile * 16;
    const int n0    = ng * 64;

    Acc8 acc[4];
#pragma unroll
    for (int t = 0; t < 4; ++t)
#pragma unroll
        for (int j = 0; j < 8; ++j) acc[t].f[j] = 0.f;

    const __bf16* arowp = Xb + (size_t)(m0 + lrow) * En + (hi ? 16 : 0);
    for (int e0 = 0; e0 < En; e0 += 32) {
        // A fragment: lane = M row; K contiguous 16 (+16 for hi lanes)
        Frag16 a;
        a.q[0] = *(const uint4*)(arowp + e0);
        a.q[1] = *(const uint4*)(arowp + e0 + 8);
        // B fragments: lane = N col = weight row f; K contiguous 16
        const int kb = e0 + (hi ? 16 : 0);
#pragma unroll
        for (int t = 0; t < 4; ++t) {
            const __bf16* wp = W + (size_t)(n0 + t * 16 + lrow) * En + kb;
            Frag16 bfr;
            bfr.q[0] = *(const uint4*)(wp);
            bfr.q[1] = *(const uint4*)(wp + 8);
            acc[t].v = wmma_bf16(a.v, bfr.v, acc[t].v);
        }
    }

    // ---- bias, optional 1/sqrt(E) pre-scale (Q only), store ----
    const float qscale = (mat == 0) ? 0.04419417382415922f : 1.0f;  // 1/sqrt(512)
    const int rbase = m0 + (hi ? 8 : 0);
#pragma unroll
    for (int t = 0; t < 4; ++t) {
        const int col = n0 + t * 16 + lrow;
        const float bv_ = bias[col];
        if (mat < 2) {
            __bf16* dst = (mat == 0) ? Qb : Kb;
#pragma unroll
            for (int j = 0; j < 8; ++j) {
                const int row = rbase + j;
                dst[(size_t)row * En + col] = (__bf16)((acc[t].f[j] + bv_) * qscale);
            }
        } else {
            // transposed store: Vt[b][e=col][s]
            const int bidx = m0 >> 12;                 // row / 4096
#pragma unroll
            for (int j = 0; j < 8; ++j) {
                const int s = (rbase + j) & (Sn - 1);
                Vt[(size_t)bidx * En * Sn + (size_t)col * Sn + s] =
                    (__bf16)(acc[t].f[j] + bv_);
            }
        }
    }
}

// ---------------------------------------------------------------------------
// Pass 2: flash attention, 64-key tiles.
// Block = 4 waves, owns 16 query rows.  Wave w owns E-slice [w*128, w*128+128)
// for both Q.K^T partials and the O accumulator.
// Online softmax distributed: wave w, lanes 0..3 own rows w*4 .. w*4+3.
// ---------------------------------------------------------------------------
__global__ __launch_bounds__(128, 1)
void flash_attn_kernel(const __bf16* __restrict__ Qb,
                       const __bf16* __restrict__ Kb,
                       const __bf16* __restrict__ Vt,
                       float* __restrict__ out)
{
    __shared__ alignas(16) float  sbuf[16 * KT];       // score reduce buffer
    __shared__ alignas(16) __bf16 pbuf[16 * KT];       // P tile (bf16)
    __shared__ alignas(16) float  scalebuf[16];        // per-row rescale
    __shared__ alignas(16) float  lbuf[16];            // per-row l

    const int wave = threadIdx.x >> 5;
    const int lane = threadIdx.x & 31;
    const int lrow = lane & 15;
    const bool hi  = lane >= 16;

    const int q0   = blockIdx.x * 16;          // global row in [0, BS)
    const int b    = q0 >> 12;                 // batch
    const int kr0  = b << 12;                  // first key row of this batch
    const int eb   = wave * 128;               // this wave's E-slice

    // ---- load Q fragments once (4 k-steps of 32 over the 128-wide slice) ----
    Frag16 qf[4];
#pragma unroll
    for (int kk = 0; kk < 4; ++kk) {
        const int ek = eb + kk * 32;
        const __bf16* qp = Qb + (size_t)(q0 + lrow) * En + ek + (hi ? 16 : 0);
        qf[kk].q[0] = *(const uint4*)(qp);
        qf[kk].q[1] = *(const uint4*)(qp + 8);
    }

    Acc8 o[8];
#pragma unroll
    for (int t = 0; t < 8; ++t)
#pragma unroll
        for (int j = 0; j < 8; ++j) o[t].f[j] = 0.f;

    // softmax row state: wave w, lanes 0..3 own row (wave*4 + lane)
    float m_run = -INFINITY;
    float l_run = 0.f;
    const int myrow = wave * 4 + lane;         // valid when lane < 4

    const size_t vbase = (size_t)b * En * Sn;

    for (int kt = 0; kt < Sn; kt += KT) {
        __syncthreads();                       // prior iter done with sbuf/pbuf
        *(float4*)&sbuf[threadIdx.x * 8]     = make_float4(0.f, 0.f, 0.f, 0.f);
        *(float4*)&sbuf[threadIdx.x * 8 + 4] = make_float4(0.f, 0.f, 0.f, 0.f);

        // prefetch next key tile (latency hint -> global_prefetch_b8)
        const int kt2 = kt + KT;
        if (kt2 < Sn) {
            __builtin_prefetch(Kb + (size_t)(kr0 + kt2 + lrow) * En + eb, 0, 1);
            __builtin_prefetch(Vt + vbase + (size_t)(eb + lrow) * Sn + kt2, 0, 1);
        }
        __syncthreads();

        // ---- partial scores over this wave's e-slice: S = Q . K^T ----
        Acc8 s[4];
#pragma unroll
        for (int t = 0; t < 4; ++t)
#pragma unroll
            for (int j = 0; j < 8; ++j) s[t].f[j] = 0.f;
#pragma unroll
        for (int kk = 0; kk < 4; ++kk) {
            const int ek = eb + kk * 32;
#pragma unroll
            for (int t = 0; t < 4; ++t) {      // 4 key sub-tiles of 16
                const int key = kr0 + kt + t * 16 + lrow;
                const __bf16* kp = Kb + (size_t)key * En + ek + (hi ? 16 : 0);
                Frag16 kf;
                kf.q[0] = *(const uint4*)(kp);
                kf.q[1] = *(const uint4*)(kp + 8);
                s[t].v = wmma_bf16(qf[kk].v, kf.v, s[t].v);
            }
        }
        // reduce partials across the 4 waves
        const int rowoff = hi ? 8 : 0;
#pragma unroll
        for (int t = 0; t < 4; ++t)
#pragma unroll
            for (int j = 0; j < 8; ++j)
                atomicAdd(&sbuf[(j + rowoff) * KT + t * 16 + lrow], s[t].f[j]);
        __syncthreads();

        // ---- online softmax: each wave handles 4 rows (lanes 0..3) ----
        if (lane < 4) {
            float r[KT];
            const float4* sp = (const float4*)&sbuf[myrow * KT];
#pragma unroll
            for (int i = 0; i < KT / 4; ++i) {
                float4 v4 = sp[i];
                r[i * 4 + 0] = v4.x; r[i * 4 + 1] = v4.y;
                r[i * 4 + 2] = v4.z; r[i * 4 + 3] = v4.w;
            }
            float mnew = m_run;
#pragma unroll
            for (int i = 0; i < KT; ++i) mnew = fmaxf(mnew, r[i]);
            const float sc = __expf(m_run - mnew);     // 0 on first iteration
            float sum = 0.f;
            union { uint4 q[KT / 8]; __bf16 e[KT]; } prow;
#pragma unroll
            for (int i = 0; i < KT; ++i) {
                const float pe = __expf(r[i] - mnew);
                sum += pe;
                prow.e[i] = (__bf16)pe;
            }
            l_run = l_run * sc + sum;
            m_run = mnew;
            scalebuf[myrow] = sc;
            uint4* pd = (uint4*)&pbuf[myrow * KT];
#pragma unroll
            for (int i = 0; i < KT / 8; ++i) pd[i] = prow.q[i];
        }
        __syncthreads();

        // ---- rescale O by exp(m_old - m_new) per row ----
        float scv[8];
        {
            float4 sA = *(const float4*)&scalebuf[rowoff];
            float4 sB = *(const float4*)&scalebuf[rowoff + 4];
            scv[0] = sA.x; scv[1] = sA.y; scv[2] = sA.z; scv[3] = sA.w;
            scv[4] = sB.x; scv[5] = sB.y; scv[6] = sB.z; scv[7] = sB.w;
        }
#pragma unroll
        for (int t = 0; t < 8; ++t)
#pragma unroll
            for (int j = 0; j < 8; ++j) o[t].f[j] *= scv[j];

        // ---- O += P @ V over the 64-key tile (2 k-steps of 32) ----
#pragma unroll
        for (int kk2 = 0; kk2 < 2; ++kk2) {
            // P fragment (A layout) from LDS: 32-wide window kk2*32
            Frag16 pf;
            const __bf16* pp = &pbuf[lrow * KT + kk2 * 32 + (hi ? 16 : 0)];
            pf.q[0] = *(const uint4*)(pp);
            pf.q[1] = *(const uint4*)(pp + 8);
#pragma unroll
            for (int t = 0; t < 8; ++t) {      // 8 E sub-tiles of 16
                const int ecol = eb + t * 16 + lrow;
                const __bf16* vp =
                    Vt + vbase + (size_t)ecol * Sn + kt + kk2 * 32 + (hi ? 16 : 0);
                Frag16 vf;
                vf.q[0] = *(const uint4*)(vp);
                vf.q[1] = *(const uint4*)(vp + 8);
                o[t].v = wmma_bf16(pf.v, vf.v, o[t].v);
            }
        }
    }

    __syncthreads();
    if (lane < 4) lbuf[myrow] = l_run;
    __syncthreads();

    float linv[8];
    {
        const int rowoff = hi ? 8 : 0;
#pragma unroll
        for (int j = 0; j < 8; ++j) linv[j] = 1.0f / lbuf[rowoff + j];
    }
#pragma unroll
    for (int t = 0; t < 8; ++t) {
        const int col = eb + t * 16 + lrow;
#pragma unroll
        for (int j = 0; j < 8; ++j) {
            const int row = q0 + j + (hi ? 8 : 0);
            out[(size_t)row * En + col] = o[t].f[j] * linv[j];
        }
    }
}

// ---------------------------------------------------------------------------
// Launch
// ---------------------------------------------------------------------------
extern "C" void kernel_launch(void* const* d_in, const int* in_sizes, int n_in,
                              void* d_out, int out_size, void* d_ws, size_t ws_size,
                              hipStream_t stream)
{
    (void)in_sizes; (void)n_in; (void)out_size; (void)ws_size;

    const float* x  = (const float*)d_in[0];
    const float* Wq = (const float*)d_in[1];
    const float* bq = (const float*)d_in[2];
    const float* Wk = (const float*)d_in[3];
    const float* bk = (const float*)d_in[4];
    const float* Wv = (const float*)d_in[5];
    const float* bv = (const float*)d_in[6];

    // Workspace layout (bf16 elements)
    __bf16* Qb = (__bf16*)d_ws;                        // [BS,E]   16 MB
    __bf16* Kb = Qb + (size_t)BS * En;                 // [BS,E]   16 MB
    __bf16* Vt = Kb + (size_t)BS * En;                 // [B,E,S]  16 MB
    __bf16* Xb = Vt + (size_t)BS * En;                 // [BS,E]   16 MB
    __bf16* Wb = Xb + (size_t)BS * En;                 // [3,E,E] 1.5 MB

    // Pass 0: one-time f32 -> bf16 conversion of x and weights
    cvt_bf16_kernel<<<(BS * En / 8) / 256, 256, 0, stream>>>(x, Xb);
    cvt_bf16_kernel<<<(En * En / 8) / 256, 256, 0, stream>>>(Wq, Wb + 0 * (size_t)En * En);
    cvt_bf16_kernel<<<(En * En / 8) / 256, 256, 0, stream>>>(Wk, Wb + 1 * (size_t)En * En);
    cvt_bf16_kernel<<<(En * En / 8) / 256, 256, 0, stream>>>(Wv, Wb + 2 * (size_t)En * En);

    // Pass 1: 1024 blocks x 8 waves = one 16x64 tile per wave, per matrix
    qkv_proj_kernel<<<dim3(1024, 3, 1), 256, 0, stream>>>(
        Xb, Wb, bq, bk, bv, Qb, Kb, Vt);

    // Pass 2: one block per 16 query rows
    flash_attn_kernel<<<dim3(BS / 16, 1, 1), 128, 0, stream>>>(
        Qb, Kb, Vt, (float*)d_out);
}
